// DS_HGNN_layer_77721728188418
// MI455X (gfx1250) — matrix operand order repaired and checked by reading, not compiled, not run on hardware
//
#include <hip/hip_runtime.h>

// ---------------------------------------------------------------------------
// DS-HGNN layer for MI455X (gfx1250, wave32, WMMA bf16, async-to-LDS pipeline)
// ---------------------------------------------------------------------------
typedef __attribute__((ext_vector_type(16))) __bf16 bf16x16;
typedef __attribute__((ext_vector_type(8)))  float  v8f;
typedef __attribute__((ext_vector_type(4)))  int    v4i;

#if __has_builtin(__builtin_amdgcn_global_load_async_to_lds_b128) && \
    __has_builtin(__builtin_amdgcn_s_wait_asynccnt)
#define USE_ASYNC_LDS 1
#else
#define USE_ASYNC_LDS 0
#endif

typedef __attribute__((address_space(1))) v4i gv4i;   // global int4
typedef __attribute__((address_space(3))) v4i lv4i;   // LDS int4
__device__ __forceinline__ gv4i* to_as1(const void* p) {
  return (gv4i*)(unsigned long long)p;                 // flat global == as1
}
__device__ __forceinline__ lv4i* to_as3(const void* p) {
  return (lv4i*)(unsigned int)(unsigned long long)p;   // low 32 bits = LDS offset
}

static constexpr int   BB    = 8;      // batch
static constexpr int   NV    = 1024;   // vertices
static constexpr int   CH    = 128;    // channels
static constexpr int   EED   = 1220;   // real edges (1024 kNN + 196 local)
static constexpr int   EPAD  = 1280;   // padded edge dim (mult of 256)
static constexpr int   KCAP  = 64;     // neighbor cap kept from argsort
static constexpr int   NLOC  = 196;    // local edges
static constexpr float ALPHA_V = 0.05f;
static constexpr float ALPHA_E = 0.9f;

// workspace layout (bytes)
static constexpr size_t SZ_WVE = (size_t)BB * NV * EPAD * 2;      // bf16
static constexpr size_t SZ_WEV = (size_t)BB * EPAD * NV * 2;      // bf16
static constexpr size_t SZ_E   = (size_t)BB * EPAD * CH * 4;      // f32
static constexpr size_t SZ_X   = (size_t)BB * NV * CH * 4;        // f32
static constexpr size_t SZ_ORD = (size_t)BB * NV * KCAP * 4;      // int
static constexpr size_t SZ_CNT = (size_t)BB * NV * 4;             // int
static constexpr size_t SZ_DEI = (size_t)BB * EPAD * 4;           // f32
static constexpr size_t SZ_ACC = 256 * 4;                         // f32

static constexpr size_t OFF_WVE = 0;
static constexpr size_t OFF_WEV = OFF_WVE + SZ_WVE;
static constexpr size_t OFF_EA  = OFF_WEV + SZ_WEV;
static constexpr size_t OFF_EB  = OFF_EA  + SZ_E;
static constexpr size_t OFF_XA  = OFF_EB  + SZ_E;
static constexpr size_t OFF_XB  = OFF_XA  + SZ_X;
static constexpr size_t OFF_ORD = OFF_XB  + SZ_X;
static constexpr size_t OFF_DVC = OFF_ORD + SZ_ORD;   // top-11 in-degree Dv
static constexpr size_t OFF_DVF = OFF_DVC + SZ_CNT;   // full vertex degree
static constexpr size_t OFF_DVI = OFF_DVF + SZ_CNT;   // 1/DV
static constexpr size_t OFF_DEI = OFF_DVI + SZ_CNT;   // 1/DE
static constexpr size_t OFF_ACC = OFF_DEI + SZ_DEI;
static constexpr size_t WS_TOTAL = OFF_ACC + SZ_ACC;

__device__ __forceinline__ unsigned short f2bf(float f) {
  unsigned int u = __float_as_uint(f);
  u += 0x7FFFu + ((u >> 16) & 1u);          // round-to-nearest-even
  return (unsigned short)(u >> 16);
}

// ---------------------------------------------------------------------------
__global__ void k_zero(uint4* p, size_t n16) {
  size_t i = (size_t)blockIdx.x * blockDim.x + threadIdx.x;
  size_t st = (size_t)gridDim.x * blockDim.x;
  uint4 z; z.x = z.y = z.z = z.w = 0u;
  for (; i < n16; i += st) p[i] = z;
}

__global__ void k_copy4(const float4* __restrict__ src, float4* __restrict__ dst, size_t n) {
  size_t i = (size_t)blockIdx.x * blockDim.x + threadIdx.x;
  if (i < n) dst[i] = src[i];
}

// ---------------------------------------------------------------------------
// Per-point distance row + bitonic top-64 (one workgroup per (point,batch))
// ---------------------------------------------------------------------------
__global__ __launch_bounds__(256) void k_topk(const float* __restrict__ X,
                                              int* __restrict__ order) {
  const int i = blockIdx.x, b = blockIdx.y, t = threadIdx.x;
  __shared__ float xi[CH];
  __shared__ float dist[NV];
  __shared__ int   idx[NV];
  if (t < CH) xi[t] = X[((size_t)b * NV + i) * CH + t];
  __syncthreads();
  for (int j = t; j < NV; j += 256) {
    const float* xj = &X[((size_t)b * NV + j) * CH];
    float d = 0.f;
    #pragma unroll 4
    for (int c = 0; c < CH; c += 4) {
      float4 v = *(const float4*)(xj + c);
      float d0 = xi[c] - v.x, d1 = xi[c + 1] - v.y;
      float d2 = xi[c + 2] - v.z, d3 = xi[c + 3] - v.w;
      d += d0 * d0 + d1 * d1 + d2 * d2 + d3 * d3;
    }
    dist[j] = d; idx[j] = j;
  }
  __syncthreads();
  // bitonic sort ascending (tie-break on index -> mimics stable argsort)
  for (int k = 2; k <= NV; k <<= 1) {
    for (int jj = k >> 1; jj > 0; jj >>= 1) {
      for (int e = t; e < NV; e += 256) {
        int l = e ^ jj;
        if (l > e) {
          float de = dist[e], dl = dist[l];
          int ie = idx[e], il = idx[l];
          bool lless = (dl < de) || (dl == de && il < ie);
          bool doswap = ((e & k) == 0) ? lless : !lless;
          if (doswap) { dist[e] = dl; dist[l] = de; idx[e] = il; idx[l] = ie; }
        }
      }
      __syncthreads();
    }
  }
  if (t < KCAP) order[(((size_t)b * NV) + i) * KCAP + t] = idx[t];
}

// Dv = #edges whose top-(k+1)=11 list contains vertex v
__global__ void k_count_dv(const int* __restrict__ order, int* __restrict__ DvCnt) {
  int g = blockIdx.x * blockDim.x + threadIdx.x;
  if (g >= BB * NV) return;
  const int* o = &order[(size_t)g * KCAP];
  int b = g >> 10;
  #pragma unroll
  for (int t = 0; t < 11; ++t) atomicAdd(&DvCnt[(b << 10) + o[t]], 1);
}

// full degrees from the Dv-sized kNN columns; DE(kNN edge e) = Dv[e]
__global__ void k_count_dvfull(const int* __restrict__ order, const int* __restrict__ DvCnt,
                               int* __restrict__ DVfull, float* __restrict__ DEinv) {
  int g = blockIdx.x * blockDim.x + threadIdx.x;
  if (g >= BB * NV) return;
  int b = g >> 10, e = g & (NV - 1);
  int kv = min(DvCnt[g], KCAP); if (kv < 1) kv = 1;
  DEinv[(size_t)b * EPAD + e] = 1.0f / (float)kv;
  const int* o = &order[(size_t)g * KCAP];
  for (int t = 0; t < kv; ++t) atomicAdd(&DVfull[(b << 10) + o[t]], 1);
}

// local_H row/col degree contributions, then invert DV
__global__ void k_localdeg(const float* __restrict__ localH, int* __restrict__ DVfull,
                           float* __restrict__ DVinv, float* __restrict__ DEinv) {
  int g = blockIdx.x * blockDim.x + threadIdx.x;
  if (g >= BB * NV) return;
  int b = g >> 10, q = g & (NV - 1);
  float s = 0.f;
  for (int e = 0; e < NLOC; ++e) s += localH[q * NLOC + e];
  int tot = DVfull[g] + (int)(s + 0.5f);
  DVfull[g] = tot;
  DVinv[g] = 1.0f / (float)max(tot, 1);
  if (q < NLOC) {
    float cs = 0.f;
    for (int v = 0; v < NV; ++v) cs += localH[v * NLOC + q];
    DEinv[(size_t)b * EPAD + NV + q] = 1.0f / fmaxf(cs, 1.0f);
  }
}

// scatter normalized incidence into bf16 W_ve (N x EPAD) and W_ev (EPAD x N)
__global__ void k_scatter(const int* __restrict__ order, const int* __restrict__ DvCnt,
                          const float* __restrict__ localH,
                          const float* __restrict__ DVinv, const float* __restrict__ DEinv,
                          unsigned short* __restrict__ Wve, unsigned short* __restrict__ Wev) {
  int g = blockIdx.x * blockDim.x + threadIdx.x;
  if (g >= BB * NV) return;
  int b = g >> 10, q = g & (NV - 1);
  // kNN edge q
  int kv = min(DvCnt[g], KCAP); if (kv < 1) kv = 1;
  unsigned short deib = f2bf(DEinv[(size_t)b * EPAD + q]);
  const int* o = &order[(size_t)g * KCAP];
  for (int t = 0; t < kv; ++t) {
    int v = o[t];
    Wve[((size_t)b * NV + v) * EPAD + q] = f2bf(DVinv[(b << 10) + v]);
    Wev[((size_t)b * EPAD + q) * NV + v] = deib;
  }
  // local edges touching vertex q
  unsigned short dvib = f2bf(DVinv[g]);
  for (int e = 0; e < NLOC; ++e) {
    if (localH[q * NLOC + e] > 0.5f) {
      int ed = NV + e;
      Wve[((size_t)b * NV + q) * EPAD + ed] = dvib;
      Wev[((size_t)b * EPAD + ed) * NV + q] = f2bf(DEinv[(size_t)b * EPAD + ed]);
    }
  }
}

// ---------------------------------------------------------------------------
// WMMA bf16 GEMM: 256x128 macro-tile, 8 waves each 32x128 (2x8 16x16 frags),
// K-step 32, double-buffered LDS; A staged via GLOBAL_LOAD_ASYNC_TO_LDS_B128
// when available (bf16 operand = pure copy), B converted f32->bf16 by VALU.
//  ABF : A is pre-converted bf16 (row-major, stride lda); else fp32 (stride K)
//  BT  : B is the 128x128 weight laid out (n,k); else activation (k,n)
//  EPI : 0 -> Dst = alpha*Acc + beta*Old        (diffusion lerp)
//        1 -> Dst = Old + relu(Acc + bias[col]) (MLP residual)
// ---------------------------------------------------------------------------
template <bool ABF, bool BT, int EPI>
__global__ __launch_bounds__(256) void k_gemm(
    const void* __restrict__ A_, int lda, size_t aStride,
    const float* __restrict__ Bm, size_t bStride,
    const float* __restrict__ Old, size_t oStride,
    const float* __restrict__ bias,
    float* __restrict__ Dst,
    int K, float alpha, float beta) {
  constexpr int ALD = 40;                               // padded LDS row (bf16)
  __shared__ alignas(16) unsigned short As[2][256 * ALD];   // A tiles 256x32
  __shared__ alignas(16) unsigned short Bt[2][128 * ALD];   // B tiles (n,k)
  const int t = threadIdx.x;
  const int bidx = blockIdx.y;
  const int mTile = blockIdx.x * 256;
  const float* Bp = Bm + bidx * bStride;
  const float* Op = Old + bidx * oStride;
  float* Dp = Dst + bidx * oStride;
  const unsigned short* Apb =
      (const unsigned short*)A_ + bidx * aStride + (size_t)mTile * lda;
  const float* Apf = (const float*)A_ + bidx * aStride + (size_t)mTile * K;

  const int lane = t & 31, wid = t >> 5;
  const int wrow = wid * 32;                 // wave's 32-row strip
  const int lrow = lane & 15, lhi = lane >> 4;

  v8f zero = {0.f, 0.f, 0.f, 0.f, 0.f, 0.f, 0.f, 0.f};
  v8f acc[2][8];
  #pragma unroll
  for (int i = 0; i < 2; ++i)
    #pragma unroll
    for (int j = 0; j < 8; ++j) acc[i][j] = zero;

  union F16 { bf16x16 v; uint4 q[2]; };

  auto fillA = [&](int buf, int k0) {
    if (ABF) {
#if USE_ASYNC_LDS
      #pragma unroll
      for (int r = 0; r < 4; ++r) {                // 1024 x b128 async copies
        int id = t + r * 256;
        int row = id >> 2, seg = (id & 3) * 8;     // 4 x 16B per 64B row
        __builtin_amdgcn_global_load_async_to_lds_b128(
            to_as1(&Apb[(size_t)row * lda + k0 + seg]),
            to_as3(&As[buf][row * ALD + seg]), 0, 0);
      }
#else
      #pragma unroll
      for (int r = 0; r < 16; ++r) {               // 4096 dword copies
        int id = t + r * 256;
        int row = id >> 4, c2 = (id & 15) * 2;
        *(unsigned int*)&As[buf][row * ALD + c2] =
            *(const unsigned int*)&Apb[(size_t)row * lda + k0 + c2];
      }
#endif
    } else {
      #pragma unroll
      for (int r = 0; r < 32; ++r) {               // 8192 f32 -> bf16
        int id = t + r * 256;
        int row = id >> 5, c = id & 31;
        As[buf][row * ALD + c] = f2bf(Apf[(size_t)row * K + k0 + c]);
      }
    }
  };
  auto fillB = [&](int buf, int k0) {
    if (BT) {                                      // weight (n,k)
      #pragma unroll
      for (int r = 0; r < 16; ++r) {
        int id = t + r * 256;
        int n = id >> 5, k = id & 31;
        Bt[buf][n * ALD + k] = f2bf(Bp[(size_t)n * K + k0 + k]);
      }
    } else {                                       // activation (k, 128)
      #pragma unroll
      for (int r = 0; r < 16; ++r) {
        int id = t + r * 256;
        int k = id >> 7, n = id & 127;
        Bt[buf][n * ALD + k] = f2bf(Bp[(size_t)(k0 + k) * CH + n]);
      }
    }
  };
  auto compute = [&](int buf) {
    // A 16x32 frag: lanes 0-15 row M, K 0..7 & 16..23; lanes 16-31 K 8..15 & 24..31
    F16 fa0, fa1;
    {
      const uint4* p = (const uint4*)&As[buf][(wrow + lrow) * ALD + lhi * 8];
      fa0.q[0] = p[0]; fa0.q[1] = p[2];
      const uint4* p2 = (const uint4*)&As[buf][(wrow + 16 + lrow) * ALD + lhi * 8];
      fa1.q[0] = p2[0]; fa1.q[1] = p2[2];
    }
    #pragma unroll
    for (int j = 0; j < 8; ++j) {
      // B 32x16 frag: lane = N col; lanes 0-15 K 0..7 & 16..23; lanes 16-31 K 8..15 & 24..31
      F16 fb;
      const uint4* p = (const uint4*)&Bt[buf][(j * 16 + lrow) * ALD + lhi * 8];
      fb.q[0] = p[0]; fb.q[1] = p[2];
      acc[0][j] = __builtin_amdgcn_wmma_f32_16x16x32_bf16(
          false, fa0.v, false, fb.v, (short)0, acc[0][j], false, false);
      acc[1][j] = __builtin_amdgcn_wmma_f32_16x16x32_bf16(
          false, fa1.v, false, fb.v, (short)0, acc[1][j], false, false);
    }
  };

  const int nK = K >> 5;
  fillA(0, 0);
  fillB(0, 0);
#if USE_ASYNC_LDS
  if (ABF) __builtin_amdgcn_s_wait_asynccnt(0);
#endif
  __syncthreads();
  for (int kt = 0; kt < nK; ++kt) {
    const int cur = kt & 1, nxt = cur ^ 1;
    if (kt + 1 < nK) {                 // stage next tile while WMMAs run
      fillA(nxt, (kt + 1) << 5);
      fillB(nxt, (kt + 1) << 5);
    }
    compute(cur);
#if USE_ASYNC_LDS
    if (ABF) __builtin_amdgcn_s_wait_asynccnt(0);
#endif
    __syncthreads();
  }

  // epilogue: C/D layout -> reg r, lane: M = r + 8*(lane>>4), N = lane&15
  #pragma unroll
  for (int i = 0; i < 2; ++i) {
    #pragma unroll
    for (int j = 0; j < 8; ++j) {
      int rowb = mTile + wrow + i * 16 + lhi * 8;
      int col = j * 16 + lrow;
      #pragma unroll
      for (int r = 0; r < 8; ++r) {
        size_t off = (size_t)(rowb + r) * CH + col;
        float a = acc[i][j][r];
        if (EPI == 0) {
          Dp[off] = alpha * a + beta * Op[off];
        } else {
          Dp[off] = Op[off] + fmaxf(a + bias[col], 0.f);
        }
      }
    }
  }
}

// ---------------------------------------------------------------------------
// per-node BN: mean/var over (B, C) = 1024 values for each node (one WG/node)
// ---------------------------------------------------------------------------
__global__ __launch_bounds__(256) void k_bn_node(float* __restrict__ Z, int LD) {
  const int n = blockIdx.x, t = threadIdx.x;
  __shared__ float sb[256], sb2[256];
  __shared__ float stats[2];
  float s = 0.f, s2 = 0.f;
  #pragma unroll
  for (int r = 0; r < 4; ++r) {
    int id = t + r * 256;
    int bb = id >> 7, c = id & 127;
    float v = Z[((size_t)bb * LD + n) * CH + c];
    s += v; s2 += v * v;
  }
  sb[t] = s; sb2[t] = s2;
  __syncthreads();
  for (int o = 128; o > 0; o >>= 1) {
    if (t < o) { sb[t] += sb[t + o]; sb2[t] += sb2[t + o]; }
    __syncthreads();
  }
  if (t == 0) {
    float m = sb[0] * (1.0f / 1024.0f);
    float var = sb2[0] * (1.0f / 1024.0f) - m * m;
    stats[0] = m; stats[1] = rsqrtf(var + 1e-5f);
  }
  __syncthreads();
  float m = stats[0], rs = stats[1];
  #pragma unroll
  for (int r = 0; r < 4; ++r) {
    int id = t + r * 256;
    int bb = id >> 7, c = id & 127;
    size_t off = ((size_t)bb * LD + n) * CH + c;
    Z[off] = (Z[off] - m) * rs;
  }
}

// final per-channel stats over all (B,N) rows
__global__ __launch_bounds__(256) void k_redfinal(const float* __restrict__ Xd,
                                                  float* __restrict__ accum) {
  const int t = threadIdx.x, c = t & 127, half = t >> 7;
  __shared__ float sb[256], sb2[256];
  float s = 0.f, s2 = 0.f;
  int row0 = blockIdx.x * 64;
  for (int r = half; r < 64; r += 2) {
    float v = Xd[(size_t)(row0 + r) * CH + c];
    s += v; s2 += v * v;
  }
  sb[t] = s; sb2[t] = s2;
  __syncthreads();
  if (t < 128) {
    atomicAdd(&accum[c], sb[t] + sb[t + 128]);
    atomicAdd(&accum[128 + c], sb2[t] + sb2[t + 128]);
  }
}

__global__ void k_final(const float* __restrict__ Xd, const float* __restrict__ Xin,
                        const float* __restrict__ accum,
                        const float* __restrict__ bnw, const float* __restrict__ bnb,
                        float* __restrict__ out) {
  size_t g = (size_t)blockIdx.x * blockDim.x + threadIdx.x;
  if (g >= (size_t)BB * NV * CH) return;
  int c = (int)(g & 127);
  const float inv = 1.0f / (float)(BB * NV);
  float m = accum[c] * inv;
  float var = accum[128 + c] * inv - m * m;
  float xn = (Xd[g] - m) * rsqrtf(var + 1e-5f) * bnw[c] + bnb[c];
  out[g] = fmaxf(xn, 0.f) + Xin[g];
}

// ---------------------------------------------------------------------------
extern "C" void kernel_launch(void* const* d_in, const int* in_sizes, int n_in,
                              void* d_out, int out_size, void* d_ws, size_t ws_size,
                              hipStream_t stream) {
  const float* Xin    = (const float*)d_in[0];
  const float* localH = (const float*)d_in[1];
  const float* tv_w   = (const float*)d_in[2];
  const float* tv_b   = (const float*)d_in[3];
  const float* te_w   = (const float*)d_in[4];
  const float* te_b   = (const float*)d_in[5];
  const float* bn_w   = (const float*)d_in[6];
  const float* bn_b   = (const float*)d_in[7];

  char* ws = (char*)d_ws;
  unsigned short* Wve = (unsigned short*)(ws + OFF_WVE);
  unsigned short* Wev = (unsigned short*)(ws + OFF_WEV);
  float* Ea = (float*)(ws + OFF_EA);
  float* Eb = (float*)(ws + OFF_EB);
  float* Xa = (float*)(ws + OFF_XA);
  float* Xb = (float*)(ws + OFF_XB);
  int*   order  = (int*)(ws + OFF_ORD);
  int*   DvCnt  = (int*)(ws + OFF_DVC);
  int*   DVfull = (int*)(ws + OFF_DVF);
  float* DVinv  = (float*)(ws + OFF_DVI);
  float* DEinv  = (float*)(ws + OFF_DEI);
  float* accum  = (float*)(ws + OFF_ACC);

  // 1) fresh scratch every launch (graph-replay safe)
  k_zero<<<2048, 256, 0, stream>>>((uint4*)ws, WS_TOTAL / 16);
  k_copy4<<<(BB * NV * CH / 4 + 255) / 256, 256, 0, stream>>>(
      (const float4*)Xin, (float4*)Xa, (size_t)BB * NV * CH / 4);

  // 2) build data-dependent hypergraph operators
  k_topk<<<dim3(NV, BB), 256, 0, stream>>>(Xin, order);
  k_count_dv<<<(BB * NV + 255) / 256, 256, 0, stream>>>(order, DvCnt);
  k_count_dvfull<<<(BB * NV + 255) / 256, 256, 0, stream>>>(order, DvCnt, DVfull, DEinv);
  k_localdeg<<<(BB * NV + 255) / 256, 256, 0, stream>>>(localH, DVfull, DVinv, DEinv);
  k_scatter<<<(BB * NV + 255) / 256, 256, 0, stream>>>(order, DvCnt, localH, DVinv, DEinv,
                                                       Wve, Wev);

  float *Xc = Xa, *Xn = Xb, *Ec = Ea, *En = Eb;
  const size_t sWve = (size_t)NV * EPAD, sWev = (size_t)EPAD * NV;
  const size_t sX = (size_t)NV * CH, sE = (size_t)EPAD * CH;

  // 3) E = W_ev @ X   (alpha=1, beta=0; Old==Dst is safe: same-thread RAW only)
  k_gemm<true, false, 0><<<dim3(EPAD / 256, BB), 256, 0, stream>>>(
      Wev, NV, sWev, Xc, sX, Ec, sE, nullptr, Ec, NV, 1.0f, 0.0f);

  // 4) diffusion
  for (int it = 0; it < 40; ++it) {
    if (it % 20 == 0) {
      k_gemm<false, true, 1><<<dim3(NV / 256, BB), 256, 0, stream>>>(
          Xc, CH, sX, tv_w, 0, Xc, sX, tv_b, Xn, CH, 1.0f, 0.0f);
      { float* tmp = Xc; Xc = Xn; Xn = tmp; }
      k_gemm<false, true, 1><<<dim3(EPAD / 256, BB), 256, 0, stream>>>(
          Ec, CH, sE, te_w, 0, Ec, sE, te_b, En, CH, 1.0f, 0.0f);
      { float* tmp = Ec; Ec = En; En = tmp; }
      k_bn_node<<<NV, 256, 0, stream>>>(Xc, NV);
      k_bn_node<<<EED, 256, 0, stream>>>(Ec, EPAD);
    }
    // X' = (1-av)X + av*(Wve @ E);  E' = (1-ae)E + ae*(Wev @ X_old)
    k_gemm<true, false, 0><<<dim3(NV / 256, BB), 256, 0, stream>>>(
        Wve, EPAD, sWve, Ec, sE, Xc, sX, nullptr, Xn, EPAD, ALPHA_V, 1.0f - ALPHA_V);
    k_gemm<true, false, 0><<<dim3(EPAD / 256, BB), 256, 0, stream>>>(
        Wev, NV, sWev, Xc, sX, Ec, sE, nullptr, En, NV, ALPHA_E, 1.0f - ALPHA_E);
    { float* tmp = Xc; Xc = Xn; Xn = tmp; }
    { float* tmp = Ec; Ec = En; En = tmp; }
  }

  // 5) final channel BN + relu + residual
  k_redfinal<<<BB * NV / 64, 256, 0, stream>>>(Xc, accum);
  k_final<<<(BB * NV * CH + 255) / 256, 256, 0, stream>>>(Xc, Xin, accum, bn_w, bn_b,
                                                          (float*)d_out);
}
